// metaLinearLayer_4561255268487
// MI455X (gfx1250) — matrix operand order repaired and checked by reading, not compiled, not run on hardware
//
#include <hip/hip_runtime.h>
#include <hip/hip_bf16.h>
#include <cstdint>

// Problem constants (match reference)
#define B_    8
#define S_    4096
#define DIN   2048
#define DOUT  2048
#define P_    256

typedef __attribute__((ext_vector_type(16))) __bf16 v16bf;
typedef __attribute__((ext_vector_type(8)))  __bf16 v8bf;
typedef __attribute__((ext_vector_type(4)))  __bf16 v4bf;
typedef __attribute__((ext_vector_type(16))) float  v16f;
typedef __attribute__((ext_vector_type(4)))  float  v4fv;
typedef __attribute__((ext_vector_type(8)))  float  v8f;

union BFVec {            // one WMMA 16-bit operand (8 VGPRs)
  v16bf v;
  uint4 q[2];
};
union F16x {             // 16 floats staged for conversion
  v16f v;
  float4 f4[4];
};

// ---------------- Prologue kernel 1: left f32 -> bf16 (same layout) --------
__global__ void cvt_left_kernel(const float* __restrict__ left,
                                __bf16* __restrict__ leftb) {
  int i = (blockIdx.x * 256 + threadIdx.x) * 8;
  union { v8bf v; uint4 q; } o;
  float4 f0 = *(const float4*)(left + i);
  float4 f1 = *(const float4*)(left + i + 4);
  v16f fv;  // only low 8 used
  fv[0]=f0.x; fv[1]=f0.y; fv[2]=f0.z; fv[3]=f0.w;
  fv[4]=f1.x; fv[5]=f1.y; fv[6]=f1.z; fv[7]=f1.w;
  v4fv a = {fv[0], fv[1], fv[2], fv[3]};
  v4fv b = {fv[4], fv[5], fv[6], fv[7]};
  union { v4bf v; uint2 u; } lo, hi;
  lo.v = __builtin_convertvector(a, v4bf);
  hi.v = __builtin_convertvector(b, v4bf);
  o.q = make_uint4(lo.u.x, lo.u.y, hi.u.x, hi.u.y);
  *(uint4*)(leftb + i) = o.q;
}

// ------------- Prologue kernel 2: right [P,DOUT] -> rightT bf16 [DOUT,P] ---
__global__ void cvt_right_t_kernel(const float* __restrict__ right,
                                   __bf16* __restrict__ rightT) {
  __shared__ float tile[32][33];
  int tx = threadIdx.x & 31;
  int ty = threadIdx.x >> 5;          // 0..7
  int n0 = blockIdx.x * 32;           // along DOUT
  int p0 = blockIdx.y * 32;           // along P
#pragma unroll
  for (int j = 0; j < 32; j += 8)
    tile[ty + j][tx] = right[(size_t)(p0 + ty + j) * DOUT + n0 + tx];
  __syncthreads();
#pragma unroll
  for (int j = 0; j < 32; j += 8)
    rightT[(size_t)(n0 + ty + j) * P_ + p0 + tx] = (__bf16)tile[tx][ty + j];
}

// ------------- Prologue kernel 3: meta_bias[b][n] = meta@bias_w + bias_b ---
__global__ void meta_bias_kernel(const float* __restrict__ meta,
                                 const float* __restrict__ bias_w,
                                 const float* __restrict__ bias_b,
                                 float* __restrict__ mb) {
  int i = blockIdx.x * 256 + threadIdx.x;   // 0 .. B_*DOUT-1
  int b = i >> 11;                          // /2048 (uniform per block)
  int n = i & (DOUT - 1);
  float acc = bias_b[n];
#pragma unroll 4
  for (int p = 0; p < P_; ++p)
    acc += meta[b * P_ + p] * bias_w[(size_t)p * DOUT + n];
  mb[i] = acc;
}

// ---------------- load helpers ---------------------------------------------
__device__ __forceinline__ void loadA_g(F16x& d, const float* xrow, int k0, int kh) {
  // A 16x32 in contiguous k order (B uses the same): lane half kh holds K [k0+16kh,+16)
  const float* p0 = xrow + k0 + 16 * kh;
  const float* p1 = xrow + k0 + 16 * kh + 8;
  d.f4[0] = *(const float4*)p0;
  d.f4[1] = *(const float4*)(p0 + 4);
  d.f4[2] = *(const float4*)p1;
  d.f4[3] = *(const float4*)(p1 + 4);
}
__device__ __forceinline__ void loadB_contig(BFVec& d, const __bf16* p) {
  // B 32x16 layout: lane = column, 16 contiguous bf16 along K
  d.q[0] = *(const uint4*)p;
  d.q[1] = *(const uint4*)(p + 8);
}

// ---------------- Main fused kernel ----------------------------------------
// grid: B_*(S_/128) workgroups, 256 threads (8 waves). Wave w owns rows
// [16w, 16w+16) of the 128-row tile. Stage-1 result H (128 x 256 bf16) is
// kept in LDS (row stride 264 bf16 = 528 B so rows spread across banks).
__global__ __launch_bounds__(256) void meta_linear_fused_kernel(
    const float* __restrict__ x, const float* __restrict__ meta,
    const __bf16* __restrict__ leftb, const __bf16* __restrict__ rightT,
    const float* __restrict__ mbias, float* __restrict__ out) {
  constexpr int HS = P_ + 8;  // 264 (rows 16B-aligned, bank-staggered)
  extern __shared__ __bf16 hs[];            // 128 * HS bf16 = 67584 B

  const int bid  = blockIdx.x;
  const int b    = bid >> 5;            // batch
  const int s0   = (bid & 31) * 128;    // row tile start
  const int lane = threadIdx.x & 31;
  const int wave = threadIdx.x >> 5;    // 0..7
  const int ml   = lane & 15;           // row (A) / col (B) within 16-tile
  const int kh   = lane >> 4;           // lane-half along K

  const v8f vzero = {0.f, 0.f, 0.f, 0.f, 0.f, 0.f, 0.f, 0.f};

  // ===================== Phase 1: H = (X @ left^T) * meta =================
  const float* xrow = x + (size_t)(b * S_ + s0 + 16 * wave + ml) * DIN;
  const __bf16* lrow = leftb + (size_t)ml * DIN + 16 * kh;

  float ms[16];
#pragma unroll
  for (int t = 0; t < 16; ++t) ms[t] = meta[b * P_ + 16 * t + ml];

  v8f acc[16];
#pragma unroll
  for (int t = 0; t < 16; ++t) acc[t] = vzero;

  F16x afc;                 // current A chunk (f32), prefetched
  loadA_g(afc, xrow, 0, kh);

  for (int k0 = 0; k0 < DIN; k0 += 32) {
    F16x afn = afc;
    if (k0 + 32 < DIN) loadA_g(afn, xrow, k0 + 32, kh);   // prefetch next k

    BFVec a;
    a.v = __builtin_convertvector(afc.v, v16bf);          // packed bf16 cvt

    BFVec bb[2];
    loadB_contig(bb[0], lrow + k0);                       // tile t=0
#pragma unroll
    for (int t = 0; t < 16; ++t) {
      if (t < 15)                                         // double-buffer B
        loadB_contig(bb[(t + 1) & 1], lrow + (size_t)(16 * (t + 1)) * DIN + k0);
      acc[t] = __builtin_amdgcn_wmma_f32_16x16x32_bf16(
          false, a.v, false, bb[t & 1].v, (short)0, acc[t], false, false);
    }
    afc = afn;
  }

  // scale by meta, convert to bf16, park H in LDS
#pragma unroll
  for (int t = 0; t < 16; ++t) {
    const int p = 16 * t + ml;
#pragma unroll
    for (int r = 0; r < 8; ++r) {
      const int m = 16 * wave + 8 * kh + r;   // C/D layout: m = r + 8*(lane/16)
      hs[m * HS + p] = (__bf16)(acc[t][r] * ms[t]);
    }
  }
  __syncthreads();

  // ===================== Phase 2: out = H @ right + meta_bias =============
  const int arow = (16 * wave + ml) * HS + 16 * kh;
  const __bf16* rbase = rightT + (size_t)ml * P_ + 16 * kh;
  float* orow = out + (size_t)(b * S_ + s0) * DOUT;

  for (int n0 = 0; n0 < DOUT; n0 += 64) {
    v8f oac[4];
#pragma unroll
    for (int tt = 0; tt < 4; ++tt) oac[tt] = vzero;

    BFVec bb[2];
    loadB_contig(bb[0], rbase + (size_t)n0 * P_);         // (k=0, tt=0)
#pragma unroll
    for (int ki = 0; ki < P_ / 32; ++ki) {
      const int k0 = ki * 32;
      // A from LDS: two ds_load_b128 per lane (DScnt, independent of loads)
      BFVec a;
      a.q[0] = *(const uint4*)&hs[arow + k0];
      a.q[1] = *(const uint4*)&hs[arow + k0 + 8];
#pragma unroll
      for (int tt = 0; tt < 4; ++tt) {
        const int l = ki * 4 + tt;
        if (l < 31) {                                     // double-buffer B
          const int ln = l + 1, ttn = ln & 3, kn = (ln >> 2) * 32;
          loadB_contig(bb[ln & 1],
                       rbase + (size_t)(n0 + 16 * ttn) * P_ + kn);
        }
        oac[tt] = __builtin_amdgcn_wmma_f32_16x16x32_bf16(
            false, a.v, false, bb[l & 1].v, (short)0, oac[tt], false, false);
      }
    }

#pragma unroll
    for (int tt = 0; tt < 4; ++tt) {
      const int n = n0 + 16 * tt + ml;
      const float mb = mbias[b * DOUT + n];
#pragma unroll
      for (int r = 0; r < 8; ++r) {
        const int m = 16 * wave + 8 * kh + r;
        orow[(size_t)m * DOUT + n] = oac[tt][r] + mb;
      }
    }
  }
}

// ---------------------------------------------------------------------------
extern "C" void kernel_launch(void* const* d_in, const int* in_sizes, int n_in,
                              void* d_out, int out_size, void* d_ws, size_t ws_size,
                              hipStream_t stream) {
  const float* x      = (const float*)d_in[0];  // [B,S,DIN]
  const float* meta   = (const float*)d_in[1];  // [B,P]
  const float* left   = (const float*)d_in[2];  // [P,DIN]
  const float* right  = (const float*)d_in[3];  // [P,DOUT]
  const float* bias_w = (const float*)d_in[4];  // [P,DOUT]
  const float* bias_b = (const float*)d_in[5];  // [DOUT]
  float* out = (float*)d_out;

  // workspace layout (needs ~2.07 MB):
  //   [0, 1MB)      left_bf16  [P][DIN]
  //   [1MB, 2MB)    rightT_bf16[DOUT][P]
  //   [2MB, +64KB)  meta_bias  [B][DOUT] f32
  __bf16* leftb  = (__bf16*)d_ws;
  __bf16* rightT = leftb + (size_t)P_ * DIN;
  float*  mbias  = (float*)(rightT + (size_t)DOUT * P_);

  cvt_left_kernel<<<(P_ * DIN) / (256 * 8), 256, 0, stream>>>(left, leftb);
  cvt_right_t_kernel<<<dim3(DOUT / 32, P_ / 32), 256, 0, stream>>>(right, rightT);
  meta_bias_kernel<<<(B_ * DOUT) / 256, 256, 0, stream>>>(meta, bias_w, bias_b, mbias);

  const size_t lds_bytes = (size_t)128 * (P_ + 8) * sizeof(__bf16);
  meta_linear_fused_kernel<<<B_ * (S_ / 128), 256, lds_bytes, stream>>>(
      x, meta, leftb, rightT, mbias, out);
}
